// MoELayer_17059610100406
// MI455X (gfx1250) — matrix operand order repaired and checked by reading, not compiled, not run on hardware
//
#include <hip/hip_runtime.h>
#include <hip/hip_bf16.h>

// ---------------------------------------------------------------------------
// MoE SwiGLU layer for gfx1250 (MI455X).
// Matrix math on v_wmma_f32_16x16x32_bf16 (fp32 accumulate); x-gather staged
// into LDS with global_load_async_to_lds_b128 (ASYNCcnt).
// ---------------------------------------------------------------------------

typedef __attribute__((ext_vector_type(16))) __bf16 v16bf;
typedef __attribute__((ext_vector_type(8)))  float  v8f;

#define N_TOK 4096
#define DM    512
#define DF    2048
#define NE    8
#define MT    32     // tokens per block tile (2 WMMA M-subtiles)
// padded LDS row strides (in ushorts); rows stay 32B-aligned for v16bf loads
#define XP    528    // 512 + 16   (1056 B = 33*32, also 16B-aligned rows)
#define HP    2064   // 2048 + 16  (4128 B = 129*32)

static __device__ __forceinline__ unsigned short f2bf(float f) {
  union { float f; unsigned u; } v; v.f = f;
  unsigned r = v.u + 0x7FFFu + ((v.u >> 16) & 1u);   // round-to-nearest-even
  return (unsigned short)(r >> 16);
}

static __device__ __forceinline__ v8f wmma_bf16(v16bf a, v16bf b, v8f c) {
  return __builtin_amdgcn_wmma_f32_16x16x32_bf16(false, a, false, b,
                                                 (short)0, c, false, false);
}

// ---------------------------------------------------------------------------
// 1) zero output + stats + expert counters (graph-replay safe)
// ---------------------------------------------------------------------------
__global__ void init_kernel(float* __restrict__ out, int n,
                            float* __restrict__ stats, int* __restrict__ cnt) {
  int i0 = blockIdx.x * blockDim.x + threadIdx.x;
  for (int i = i0; i < n; i += gridDim.x * blockDim.x) out[i] = 0.0f;
  if (i0 < 17) stats[i0] = 0.0f;   // importance[8], load[8], zsum
  if (i0 < NE) cnt[i0] = 0;
}

// ---------------------------------------------------------------------------
// 2) fp32 -> bf16 conversion (x and all expert weights)
// ---------------------------------------------------------------------------
__global__ void convert_bf16_kernel(const float* __restrict__ src,
                                    unsigned short* __restrict__ dst, int n) {
  for (int i = blockIdx.x * blockDim.x + threadIdx.x; i < n;
       i += gridDim.x * blockDim.x)
    dst[i] = f2bf(src[i]);
}

// ---------------------------------------------------------------------------
// 3) router: logits, softmax, top-2, expert lists, loss accumulators
// ---------------------------------------------------------------------------
__global__ void router_kernel(const float* __restrict__ x,
                              const float* __restrict__ rw,
                              int* __restrict__ cnt,
                              int* __restrict__ tokL,
                              float* __restrict__ cwL,
                              float* __restrict__ stats) {
  __shared__ float srw[DM * NE];                      // 16 KB
  const int tid = threadIdx.x;
  for (int i = tid; i < DM * NE; i += blockDim.x) srw[i] = rw[i];
  __syncthreads();

  const int t = blockIdx.x * blockDim.x + tid;        // one token per thread
  const float* xr = x + (size_t)t * DM;

  float lg[NE];
  #pragma unroll
  for (int e = 0; e < NE; ++e) lg[e] = 0.0f;
  for (int d = 0; d < DM; ++d) {
    const float xv = xr[d];
    #pragma unroll
    for (int e = 0; e < NE; ++e) lg[e] += xv * srw[d * NE + e];
  }

  float zs = 0.0f;
  #pragma unroll
  for (int e = 0; e < NE; ++e) zs += lg[e] * lg[e];
  atomicAdd(&stats[16], zs);

  float mx = lg[0];
  #pragma unroll
  for (int e = 1; e < NE; ++e) mx = fmaxf(mx, lg[e]);
  float p[NE], ps = 0.0f;
  #pragma unroll
  for (int e = 0; e < NE; ++e) { p[e] = __expf(lg[e] - mx); ps += p[e]; }
  const float inv = 1.0f / ps;
  #pragma unroll
  for (int e = 0; e < NE; ++e) atomicAdd(&stats[e], p[e] * inv); // importance

  int i0 = 0; float p0 = p[0];
  #pragma unroll
  for (int e = 1; e < NE; ++e) if (p[e] > p0) { p0 = p[e]; i0 = e; }
  int i1 = (i0 == 0) ? 1 : 0; float p1 = p[i1];
  #pragma unroll
  for (int e = 0; e < NE; ++e)
    if (e != i0 && p[e] > p1) { p1 = p[e]; i1 = e; }

  atomicAdd(&stats[8 + i0], 1.0f);                    // load (top-1 one-hot)

  const float wn = 1.0f / (p0 + p1);
  int pos0 = atomicAdd(&cnt[i0], 1);
  tokL[i0 * N_TOK + pos0] = t;  cwL[i0 * N_TOK + pos0] = p0 * wn;
  int pos1 = atomicAdd(&cnt[i1], 1);
  tokL[i1 * N_TOK + pos1] = t;  cwL[i1 * N_TOK + pos1] = p1 * wn;
}

// ---------------------------------------------------------------------------
// 4) scalar losses
// ---------------------------------------------------------------------------
__global__ void loss_kernel(const float* __restrict__ stats,
                            float* __restrict__ out_losses) {
  float aux = 0.0f;
  #pragma unroll
  for (int e = 0; e < NE; ++e)
    aux += (stats[e] / (float)N_TOK) * (stats[8 + e] / (float)N_TOK);
  out_losses[0] = (float)NE * aux * 0.01f;                       // aux_loss
  out_losses[1] = stats[16] / (float)(N_TOK * NE) * 1e-4f;       // z_loss
}

// ---------------------------------------------------------------------------
// 5) grouped expert GEMMs, 32-token tiles: gate/up (bf16 WMMA), SiLU*up,
//    down proj, combine-weighted scatter. Each B fragment feeds 2 WMMAs
//    (two M-subtiles) to halve L2 weight traffic per FLOP.
//    Grid: NE experts x 128 tiles. Block: 256 threads = 8 waves.
// ---------------------------------------------------------------------------
__global__ void moe_expert_kernel(const unsigned short* __restrict__ xb,
                                  const unsigned short* __restrict__ w1b,
                                  const unsigned short* __restrict__ w3b,
                                  const unsigned short* __restrict__ w2b,
                                  const int* __restrict__ cnt,
                                  const int* __restrict__ tokL,
                                  const float* __restrict__ cwL,
                                  float* __restrict__ out) {
  extern __shared__ unsigned char smem[];
  unsigned short* xs = (unsigned short*)smem;          // MT x XP  (~33 KB)
  unsigned short* hs = xs + MT * XP;                   // MT x HP  (~129 KB)
  int*   sTok = (int*)(hs + MT * HP);
  float* sCw  = (float*)(sTok + MT);

  const int e  = blockIdx.x >> 7;                      // expert
  const int t  = blockIdx.x & 127;                     // token tile
  const int nt = cnt[e];
  if (t * MT >= nt) return;                            // uniform per block
  const int rows = (nt - t * MT < MT) ? (nt - t * MT) : MT;

  const int tid = threadIdx.x;
  if (tid < MT) {
    int g = t * MT + tid;
    if (g < nt) { sTok[tid] = tokL[e * N_TOK + g]; sCw[tid] = cwL[e * N_TOK + g]; }
    else        { sTok[tid] = 0;                   sCw[tid] = 0.0f; }
  }
  __syncthreads();

  // ---- async gather: 32 bf16 token rows -> LDS (16B chunks, ASYNCcnt) ----
  // 32 rows x 1024 B = 2048 chunks of 16 B; 8 chunks per thread.
  for (int c = tid; c < MT * (DM / 8); c += blockDim.x) {
    const int m  = c >> 6;                 // 64 chunks per row
    const int kc = (c & 63) << 3;          // ushort index within row
    const unsigned short* gp = xb + (size_t)sTok[m] * DM + kc;
    const unsigned ldsOff = (unsigned)(m * XP + kc) * 2u;  // byte offset in LDS
    asm volatile("global_load_async_to_lds_b128 %0, %1, off"
                 :: "v"(ldsOff), "v"(gp) : "memory");
  }
  asm volatile("s_wait_asynccnt 0x0" ::: "memory");
  __syncthreads();

  const int lane  = tid & 31;
  const int wid   = tid >> 5;
  const int rowA  = lane & 15;           // A: lane -> row M (subtile-relative)
  const int colA  = (lane >> 4) << 4;    // A: lane half -> K 0..15 / 16..31
  const int mBase = (lane >> 4) << 3;    // C/D: lane half -> M 0..7 / 8..15
  const int nLane = lane & 15;           // C/D: column within tile

  const unsigned short* w1e = w1b + (size_t)e * DM * DF;
  const unsigned short* w3e = w3b + (size_t)e * DM * DF;
  const unsigned short* w2e = w2b + (size_t)e * DF * DM;

  // ---- phase A: h = silu(x@w1) * (x@w3), 128 N-tiles over F=2048 ----
  for (int ft = wid; ft < DF / 16; ft += 8) {
    const int n0 = ft * 16;
    v8f g0 = {}, g1 = {}, u0 = {}, u1 = {};
    for (int k0 = 0; k0 < DM; k0 += 32) {
      v16bf a0 = *(const v16bf*)&xs[rowA * XP + colA + k0];
      v16bf a1 = *(const v16bf*)&xs[(rowA + 16) * XP + colA + k0];
      v16bf b1, b3;
      for (int s = 0; s < 16; ++s) {
        b1[s] = __builtin_bit_cast(__bf16, w1e[(size_t)(k0 + colA + s) * DF + n0 + nLane]);
        b3[s] = __builtin_bit_cast(__bf16, w3e[(size_t)(k0 + colA + s) * DF + n0 + nLane]);
      }
      g0 = wmma_bf16(a0, b1, g0);
      g1 = wmma_bf16(a1, b1, g1);
      u0 = wmma_bf16(a0, b3, u0);
      u1 = wmma_bf16(a1, b3, u1);
      if (k0 + 32 < DM) {
        __builtin_prefetch(&w1e[(size_t)(k0 + 32 + lane) * DF + n0], 0, 2);
        __builtin_prefetch(&w3e[(size_t)(k0 + 32 + lane) * DF + n0], 0, 2);
      }
    }
    #pragma unroll
    for (int r = 0; r < 8; ++r) {
      const float gv0 = g0[r];
      const float gv1 = g1[r];
      const float s0  = gv0 / (1.0f + __expf(-gv0));     // silu
      const float s1  = gv1 / (1.0f + __expf(-gv1));
      hs[(mBase + r) * HP + n0 + nLane]        = f2bf(s0 * u0[r]);
      hs[(16 + mBase + r) * HP + n0 + nLane]   = f2bf(s1 * u1[r]);
    }
  }
  __syncthreads();

  // ---- phase B: out_tile = h @ w2, 32 N-tiles over D=512, scaled scatter ----
  for (int dt = wid; dt < DM / 16; dt += 8) {
    const int n0 = dt * 16;
    v8f o0 = {}, o1 = {};
    for (int k0 = 0; k0 < DF; k0 += 32) {
      v16bf a0 = *(const v16bf*)&hs[rowA * HP + colA + k0];
      v16bf a1 = *(const v16bf*)&hs[(rowA + 16) * HP + colA + k0];
      v16bf b;
      for (int s = 0; s < 16; ++s)
        b[s] = __builtin_bit_cast(__bf16, w2e[(size_t)(k0 + colA + s) * DM + n0 + nLane]);
      o0 = wmma_bf16(a0, b, o0);
      o1 = wmma_bf16(a1, b, o1);
      if (k0 + 32 < DF)
        __builtin_prefetch(&w2e[(size_t)(k0 + 32 + lane) * DM + n0], 0, 2);
    }
    #pragma unroll
    for (int r = 0; r < 8; ++r) {
      const int m0 = mBase + r;
      const int m1 = 16 + mBase + r;
      if (m0 < rows)
        atomicAdd(&out[(size_t)sTok[m0] * DM + n0 + nLane], o0[r] * sCw[m0]);
      if (m1 < rows)
        atomicAdd(&out[(size_t)sTok[m1] * DM + n0 + nLane], o1[r] * sCw[m1]);
    }
  }
}

// ---------------------------------------------------------------------------
// launcher
// ---------------------------------------------------------------------------
extern "C" void kernel_launch(void* const* d_in, const int* in_sizes, int n_in,
                              void* d_out, int out_size, void* d_ws, size_t ws_size,
                              hipStream_t stream) {
  const float* x  = (const float*)d_in[0];   // [4,1024,512]
  const float* rw = (const float*)d_in[1];   // [512,8]
  const float* w1 = (const float*)d_in[2];   // [8,512,2048]
  const float* w2 = (const float*)d_in[3];   // [8,2048,512]
  const float* w3 = (const float*)d_in[4];   // [8,512,2048]
  float* out = (float*)d_out;                // [4,1024,512] + aux + z

  unsigned char* ws = (unsigned char*)d_ws;
  const size_t SZ_XB = (size_t)N_TOK * DM * 2;          // 4 MB
  const size_t SZ_W  = (size_t)NE * DM * DF * 2;        // 16 MB each
  unsigned short* xb   = (unsigned short*)(ws);
  unsigned short* w1b  = (unsigned short*)(ws + SZ_XB);
  unsigned short* w3b  = (unsigned short*)(ws + SZ_XB + SZ_W);
  unsigned short* w2b  = (unsigned short*)(ws + SZ_XB + 2 * SZ_W);
  unsigned char*  tail = ws + SZ_XB + 3 * SZ_W;
  int*   cnt   = (int*)(tail);
  int*   tokL  = (int*)(tail + 256);
  float* cwL   = (float*)(tail + 256 + (size_t)NE * N_TOK * 4);
  float* stats = (float*)(tail + 256 + 2 * (size_t)NE * N_TOK * 4);

  init_kernel<<<2048, 256, 0, stream>>>(out, N_TOK * DM, stats, cnt);
  convert_bf16_kernel<<<4096, 256, 0, stream>>>(x,  xb,  N_TOK * DM);
  convert_bf16_kernel<<<4096, 256, 0, stream>>>(w1, w1b, NE * DM * DF);
  convert_bf16_kernel<<<4096, 256, 0, stream>>>(w3, w3b, NE * DM * DF);
  convert_bf16_kernel<<<4096, 256, 0, stream>>>(w2, w2b, NE * DF * DM);
  router_kernel<<<N_TOK / 256, 256, 0, stream>>>(x, rw, cnt, tokL, cwL, stats);
  loss_kernel<<<1, 1, 0, stream>>>(stats, out + (size_t)N_TOK * DM);

  const size_t smem = (size_t)(MT * XP + MT * HP) * 2 + MT * 4 + MT * 4;
  moe_expert_kernel<<<NE * 128, 256, smem, stream>>>(xb, w1b, w3b, w2b,
                                                     cnt, tokL, cwL, out);
}
